// GCN_18253611008246
// MI455X (gfx1250) — compile-verified, matrix operand order checked by reading, not run on hardware
//
#include <hip/hip_runtime.h>
#include <hip/hip_bf16.h>

#define NN 50000
#define NE 800000
#define DF 128
#define DH 128
#define NC 16

typedef __attribute__((ext_vector_type(16))) __bf16    v16bf;
typedef __attribute__((ext_vector_type(8)))  __bf16    v8bf;
typedef __attribute__((ext_vector_type(8)))  float     v8f;
typedef __attribute__((ext_vector_type(4)))  float     v4f;
typedef __attribute__((ext_vector_type(16))) float     v16f;
typedef __attribute__((ext_vector_type(4)))  unsigned  u32x4;
typedef __attribute__((ext_vector_type(8)))  unsigned  u32x8;

// native RNE converts: vector fptrunc lowers to v_cvt_pk_bf16_f32 on gfx1250
static __device__ __forceinline__ __bf16 f2bf(float f) { return (__bf16)f; }

// table-order (k=8h+s, 16+8h+s-8) bf16 fragment from two 16B LDS runs (2x ds_load_b128)
static __device__ __forceinline__ v16bf frag_from_lds(const __bf16* p) {
    v8bf lo = *(const v8bf*)(p);
    v8bf hi = *(const v8bf*)(p + 16);
    return __builtin_shufflevector(lo, hi, 0,1,2,3,4,5,6,7,8,9,10,11,12,13,14,15);
}
// bf16 fragment from fp32 LDS tile: 4x ds_load_b128 + packed cvts
static __device__ __forceinline__ v16bf frag_from_f32(const float* p) {
    v4f l0 = *(const v4f*)(p);
    v4f l1 = *(const v4f*)(p + 4);
    v4f h0 = *(const v4f*)(p + 16);
    v4f h1 = *(const v4f*)(p + 20);
    v8f lo = __builtin_shufflevector(l0, l1, 0,1,2,3,4,5,6,7);
    v8f hi = __builtin_shufflevector(h0, h1, 0,1,2,3,4,5,6,7);
    v16f f = __builtin_shufflevector(lo, hi, 0,1,2,3,4,5,6,7,8,9,10,11,12,13,14,15);
    return __builtin_convertvector(f, v16bf);
}

// ---------------- degree / normalization ----------------
__global__ void deg_init_k(float* deg) {
    int i = blockIdx.x * blockDim.x + threadIdx.x;
    if (i < NN) deg[i] = 1.0f;                          // self-loop
}
__global__ void deg_acc_k(const int* __restrict__ dst, float* deg) {
    int e = blockIdx.x * blockDim.x + threadIdx.x;
    if (e < NE) atomicAdd(&deg[dst[e]], 1.0f);
}
__global__ void deg_rsqrt_k(float* deg) {
    int i = blockIdx.x * blockDim.x + threadIdx.x;
    if (i < NN) deg[i] = rsqrtf(fmaxf(deg[i], 1.0f));   // deg -> dis
}

// ---- GEMM1: H[N,128] = X[N,128] @ W1[128,128]; TDM-staged X tile ----
__global__ __launch_bounds__(256) void gemm1_k(const float* __restrict__ X,
                                               const float* __restrict__ W,
                                               float* __restrict__ H) {
    __shared__ __align__(16) float  sXf[64 * 128];      // 32 KB fp32 tile (TDM-staged)
    __shared__ __align__(16) __bf16 sWt[128 * 128];     // 32 KB W^T as [n][k]
    const int tid  = threadIdx.x;
    const int row0 = blockIdx.x * 64;

    // --- Tensor Data Mover: DMA 64x128 fp32 tile global->LDS (wave 0 issues) ---
    // D# per ISA 8.3/8.4: 2D tile 128 x 64, data_size=4B, stride 128 elem.
    // tensor_dim1 = NN-row0 so hardware zero-fills OOB tail rows.
    if (tid < 32) {
        unsigned long long ga = (unsigned long long)X + (unsigned long long)row0 * 512ull;
        unsigned ldsA  = (unsigned)(unsigned long long)(&sXf[0]);
        unsigned rleft = (unsigned)(NN - row0);
        u32x4 g0;
        g0[0] = 1u;                                              // count=1 valid descriptor
        g0[1] = ldsA;                                            // lds_addr
        g0[2] = (unsigned)ga;                                    // global_addr[31:0]
        g0[3] = (unsigned)((ga >> 32) & 0x01FFFFFFull)           // global_addr[56:32]
              | 0x80000000u;                                     // type=2 ("image")
        u32x8 g1;
        g1[0] = 0x00020000u;                                     // data_size=2 (4B), mask=0
        g1[1] = 128u << 16;                                      // tensor_dim0 = 128
        g1[2] = (rleft & 0xFFFFu) << 16;                         // tensor_dim1 lo
        g1[3] = (rleft >> 16) | (128u << 16);                    // tensor_dim1 hi | tile_dim0=128
        g1[4] = 64u;                                             // tile_dim1 = 64
        g1[5] = 128u;                                            // tensor_dim0_stride = 128
        g1[6] = 0u;
        g1[7] = 0u;
        asm volatile("tensor_load_to_lds %0, %1" :: "s"(g0), "s"(g1) : "memory");
    }

    // stage W1 transposed (fp32 -> bf16, native cvt): sWt[n*128 + k]
    #pragma unroll
    for (int j = 0; j < 16; ++j) {
        int i4 = tid + 256 * j;                         // 4096 float4 of W (row-major [k][n])
        float4 w4 = ((const float4*)W)[i4];
        int k  = i4 >> 5;
        int n0 = (i4 & 31) * 4;
        sWt[(n0 + 0) * 128 + k] = f2bf(w4.x);
        sWt[(n0 + 1) * 128 + k] = f2bf(w4.y);
        sWt[(n0 + 2) * 128 + k] = f2bf(w4.z);
        sWt[(n0 + 3) * 128 + k] = f2bf(w4.w);
    }
    __builtin_amdgcn_s_wait_tensorcnt(0);               // TDM complete (no-op waves 1..7)
    __syncthreads();

    const int wv     = tid >> 5;
    const int lane   = tid & 31;
    const int laneLo = lane & 15;
    const int laneHi = lane >> 4;
    const int mrow   = (wv & 3) * 16;                   // 4 row-groups
    const int ncol0  = (wv >> 2) * 64;                  // 2 col-groups x 4 n-tiles

    v8f acc[4];
    #pragma unroll
    for (int nt = 0; nt < 4; ++nt) acc[nt] = (v8f){0.f,0.f,0.f,0.f,0.f,0.f,0.f,0.f};

    #pragma unroll
    for (int kk = 0; kk < 4; ++kk) {
        const int kb = kk * 32 + laneHi * 8;
        v16bf a = frag_from_f32(&sXf[(mrow + laneLo) * 128 + kb]);
        #pragma unroll
        for (int nt = 0; nt < 4; ++nt) {
            const int col = ncol0 + nt * 16 + laneLo;
            v16bf b = frag_from_lds(&sWt[col * 128 + kb]);
            acc[nt] = __builtin_amdgcn_wmma_f32_16x16x32_bf16(
                false, a, false, b, (short)0, acc[nt], false, false);
        }
    }

    const int rS = mrow + laneHi * 8;
    #pragma unroll
    for (int nt = 0; nt < 4; ++nt)
        #pragma unroll
        for (int v = 0; v < 8; ++v) {
            int gr = row0 + rS + v;
            if (gr < NN) H[gr * 128 + ncol0 + nt * 16 + laneLo] = acc[nt][v];
        }
}

// ---- GEMM2: O[N,16] = H1[N,128] @ W2[128,16]; async-b128-staged tile ----
__global__ __launch_bounds__(128) void gemm2_k(const float* __restrict__ H1,
                                               const float* __restrict__ W,
                                               float* __restrict__ O) {
    __shared__ __align__(16) float  sHf[64 * 128];      // 32 KB fp32 tile (async-staged)
    __shared__ __align__(16) __bf16 sWt[16 * 128];      // 4 KB W2^T as [n][k]
    const int tid  = threadIdx.x;
    const int row0 = blockIdx.x * 64;

    const unsigned ldsBase = (unsigned)(unsigned long long)(&sHf[0]);
    #pragma unroll
    for (int j = 0; j < 16; ++j) {
        int i4 = tid + 128 * j;
        int r  = i4 >> 5;
        int grow = row0 + r;
        if (grow >= NN) grow = 0;                       // clamp: garbage row never stored
        unsigned goff = (unsigned)(grow * 32 + (i4 & 31)) * 16u;
        unsigned lds  = ldsBase + (unsigned)i4 * 16u;
        asm volatile("global_load_async_to_lds_b128 %0, %1, %2"
                     :: "v"(lds), "v"(goff), "s"(H1) : "memory");
    }
    #pragma unroll
    for (int j = 0; j < 4; ++j) {
        int i4 = tid + 128 * j;                         // 512 float4 of W2 [k][n], n=16
        float4 w4 = ((const float4*)W)[i4];
        int k  = i4 >> 2;                               // 4 float4 per row
        int n0 = (i4 & 3) * 4;
        sWt[(n0 + 0) * 128 + k] = f2bf(w4.x);
        sWt[(n0 + 1) * 128 + k] = f2bf(w4.y);
        sWt[(n0 + 2) * 128 + k] = f2bf(w4.z);
        sWt[(n0 + 3) * 128 + k] = f2bf(w4.w);
    }
    asm volatile("s_wait_asynccnt 0" ::: "memory");
    __syncthreads();

    const int wv     = tid >> 5;                        // 4 waves, 16 rows each
    const int lane   = tid & 31;
    const int laneLo = lane & 15;
    const int laneHi = lane >> 4;
    const int mrow   = wv * 16;

    v8f acc = (v8f){0.f,0.f,0.f,0.f,0.f,0.f,0.f,0.f};
    #pragma unroll
    for (int kk = 0; kk < 4; ++kk) {
        const int kb = kk * 32 + laneHi * 8;
        v16bf a = frag_from_f32(&sHf[(mrow + laneLo) * 128 + kb]);
        v16bf b = frag_from_lds(&sWt[laneLo * 128 + kb]);
        acc = __builtin_amdgcn_wmma_f32_16x16x32_bf16(
            false, a, false, b, (short)0, acc, false, false);
    }

    const int rS = mrow + laneHi * 8;
    #pragma unroll
    for (int v = 0; v < 8; ++v) {
        int gr = row0 + rS + v;
        if (gr < NN) O[gr * 16 + laneLo] = acc[v];
    }
}

// ---------------- aggregation (self-loop init + edge scatter) ----------------
__global__ void selfloop1_k(const float* __restrict__ H, const float* __restrict__ dis,
                            float* __restrict__ A) {
    int i = blockIdx.x * blockDim.x + threadIdx.x;     // N*128 threads
    if (i < NN * 128) {
        float d = dis[i >> 7];
        A[i] = H[i] * d * d;
    }
}
__global__ __launch_bounds__(256) void edge_agg1_k(const int* __restrict__ src,
                                                   const int* __restrict__ dst,
                                                   const float* __restrict__ dis,
                                                   const float* __restrict__ H,
                                                   float* __restrict__ A) {
    int e = blockIdx.x * 8 + (threadIdx.x >> 5);       // one edge per wave32
    if (e >= NE) return;
    int lane = threadIdx.x & 31;
    int s = src[e], d = dst[e];
    float w = dis[s] * dis[d];
    __builtin_prefetch(&A[d * 128], 1, 0);             // global_prefetch of dst line
    float4 hv = ((const float4*)(H + s * 128))[lane];  // 16 B/lane (512 B/wave) gather (L2-resident)
    float* out = A + d * 128 + lane * 4;
    atomicAdd(out + 0, hv.x * w);
    atomicAdd(out + 1, hv.y * w);
    atomicAdd(out + 2, hv.z * w);
    atomicAdd(out + 3, hv.w * w);
}
__global__ void bias_relu_k(float* __restrict__ A, const float* __restrict__ b1) {
    int i = blockIdx.x * blockDim.x + threadIdx.x;
    if (i < NN * 128) {
        float v = A[i] + b1[i & 127];
        A[i] = v > 0.f ? v : 0.f;
    }
}
__global__ void selfloop2_k(const float* __restrict__ O, const float* __restrict__ dis,
                            float* __restrict__ A) {
    int i = blockIdx.x * blockDim.x + threadIdx.x;     // N*16 threads
    if (i < NN * 16) {
        float d = dis[i >> 4];
        A[i] = O[i] * d * d;
    }
}
__global__ void edge_agg2_k(const int* __restrict__ src, const int* __restrict__ dst,
                            const float* __restrict__ dis, const float* __restrict__ O,
                            float* __restrict__ A) {
    int i = blockIdx.x * blockDim.x + threadIdx.x;     // E*16 threads
    if (i >= NE * 16) return;
    int e = i >> 4, f = i & 15;
    int s = src[e], d = dst[e];
    atomicAdd(&A[d * 16 + f], O[s * 16 + f] * dis[s] * dis[d]);
}
__global__ void bias_sig_k(const float* __restrict__ A, const float* __restrict__ b2,
                           float* __restrict__ out) {
    int i = blockIdx.x * blockDim.x + threadIdx.x;
    if (i < NN * 16) {
        float v = A[i] + b2[i & 15];
        out[i] = 1.0f / (1.0f + __expf(-v));
    }
}

// ---------------- launcher ----------------
extern "C" void kernel_launch(void* const* d_in, const int* in_sizes, int n_in,
                              void* d_out, int out_size, void* d_ws, size_t ws_size,
                              hipStream_t stream) {
    const float* x  = (const float*)d_in[0];
    const int*   ei = (const int*)d_in[1];
    const float* W1 = (const float*)d_in[2];
    const float* b1 = (const float*)d_in[3];
    const float* W2 = (const float*)d_in[4];
    const float* b2 = (const float*)d_in[5];
    float* out = (float*)d_out;

    const int* src = ei;
    const int* dst = ei + NE;

    char* ws = (char*)d_ws;
    const size_t MB = 1024 * 1024;
    float* dis  = (float*)(ws + 0);          //  0.2 MB  [0, 1MB)
    float* h    = (float*)(ws + 1 * MB);     // 25.6 MB  [1MB, 27MB)
    float* agg1 = (float*)(ws + 27 * MB);    // 25.6 MB  [27MB, 53MB)  (becomes h1 in place)
    float* out2 = (float*)(ws + 1 * MB);     //  3.2 MB  reuse dead h region
    float* agg2 = (float*)(ws + 6 * MB);     //  3.2 MB  reuse dead h region

    const int nblk_gemm = (NN + 63) / 64;    // 782

    deg_init_k <<<(NN + 255) / 256, 256, 0, stream>>>(dis);
    deg_acc_k  <<<(NE + 255) / 256, 256, 0, stream>>>(dst, dis);
    deg_rsqrt_k<<<(NN + 255) / 256, 256, 0, stream>>>(dis);

    gemm1_k    <<<nblk_gemm, 256, 0, stream>>>(x, W1, h);
    selfloop1_k<<<(NN * 128 + 255) / 256, 256, 0, stream>>>(h, dis, agg1);
    edge_agg1_k<<<(NE + 7) / 8, 256, 0, stream>>>(src, dst, dis, h, agg1);
    bias_relu_k<<<(NN * 128 + 255) / 256, 256, 0, stream>>>(agg1, b1);

    gemm2_k    <<<nblk_gemm, 128, 0, stream>>>(agg1, W2, out2);
    selfloop2_k<<<(NN * 16 + 255) / 256, 256, 0, stream>>>(out2, dis, agg2);
    edge_agg2_k<<<(NE * 16 + 255) / 256, 256, 0, stream>>>(src, dst, dis, out2, agg2);
    bias_sig_k <<<(NN * 16 + 255) / 256, 256, 0, stream>>>(agg2, b2, out);
}
